// SelfAttention_64948495450562
// MI455X (gfx1250) — matrix operand order repaired and checked by reading, not compiled, not run on hardware
//
#include <hip/hip_runtime.h>

typedef __attribute__((ext_vector_type(16))) __bf16 v16bf;
typedef __attribute__((ext_vector_type(8)))  __bf16 v8bf;
typedef __attribute__((ext_vector_type(8)))  float  v8f;

constexpr int BATCH = 4;
constexpr int SEQ   = 2048;
constexpr int DIN   = 1024;
constexpr int HID   = 1024;
constexpr int ROWS  = BATCH * SEQ;           // 8192
#define LN_EPS 1e-5f

union ABFrag { v16bf v; v8bf h[2]; };

// A fragment, 16x32 bf16, A row-major with leading dim lda.
// Lane<16:  e0..15 = K k0+0..15  (row M = m+lane)
// Lane>=16: e0..15 = K k0+16..31
static __device__ inline v16bf load_a_frag(const __bf16* __restrict__ A, int lda,
                                           int mBase, int k0, int lane) {
  const int half = lane >> 4;
  const __bf16* p = A + (long)(mBase + (lane & 15)) * lda + k0 + half * 16;
  ABFrag f;
  f.h[0] = *(const v8bf*)(p);
  f.h[1] = *(const v8bf*)(p + 8);
  return f.v;
}

// B fragment for B = Mrows^T: lane holds column n (= row n of Mrows), 16 contiguous K.
static __device__ inline v16bf load_b_frag_rows(const __bf16* __restrict__ Mrows, int ldb,
                                                int nBase, int k0, int lane) {
  const __bf16* p = Mrows + (long)(nBase + (lane & 15)) * ldb + k0 + ((lane >> 4) << 4);
  return *(const v16bf*)p;
}

// B fragment from pre-packed (fragment-major) buffer: one contiguous 32B load per lane.
static __device__ inline v16bf load_b_frag_packed(const __bf16* __restrict__ Bp, int ntn,
                                                  int kt, int nt16, int lane) {
  return *(const v16bf*)(Bp + ((((long)kt * ntn + nt16) * 32) + lane) * 16);
}

static __device__ inline v8f wmma_bf16(v16bf a, v16bf b, v8f c) {
  return __builtin_amdgcn_wmma_f32_16x16x32_bf16(false, a, false, b, (short)0, c,
                                                 false, false);
}

// ---------------- LayerNorm: f32 -> bf16, one block per row of 1024 ----------------
__global__ void __launch_bounds__(256) ln_kernel(const float* __restrict__ x,
                                                 const float* __restrict__ w,
                                                 const float* __restrict__ b,
                                                 __bf16* __restrict__ xn) {
  __shared__ float red[256];
  const int row = blockIdx.x;
  const int t = threadIdx.x;
  const float4 xv = *(const float4*)(x + (long)row * DIN + t * 4);
  red[t] = xv.x + xv.y + xv.z + xv.w;
  __syncthreads();
  for (int o = 128; o > 0; o >>= 1) { if (t < o) red[t] += red[t + o]; __syncthreads(); }
  const float mu = red[0] * (1.0f / DIN);
  __syncthreads();
  const float d0 = xv.x - mu, d1 = xv.y - mu, d2 = xv.z - mu, d3 = xv.w - mu;
  red[t] = d0 * d0 + d1 * d1 + d2 * d2 + d3 * d3;
  __syncthreads();
  for (int o = 128; o > 0; o >>= 1) { if (t < o) red[t] += red[t + o]; __syncthreads(); }
  const float inv = rsqrtf(red[0] * (1.0f / DIN) + LN_EPS);
  const float4 wv = *(const float4*)(w + t * 4);
  const float4 bv = *(const float4*)(b + t * 4);
  __bf16* o = xn + (long)row * DIN + t * 4;
  o[0] = (__bf16)(d0 * inv * wv.x + bv.x);
  o[1] = (__bf16)(d1 * inv * wv.y + bv.y);
  o[2] = (__bf16)(d2 * inv * wv.z + bv.z);
  o[3] = (__bf16)(d3 * inv * wv.w + bv.w);
}

// ---------------- Pack a row-major [K x Nn] matrix into B-fragment layout ----------------
template <typename SrcT>
__global__ void __launch_bounds__(256) pack_b_kernel(const SrcT* __restrict__ src,
                                                     __bf16* __restrict__ dst, int Nn,
                                                     long srcBatchStride, long dstBatchStride) {
  const long bb = blockIdx.y;
  src += bb * srcBatchStride;
  dst += bb * dstBatchStride;
  const long i = (long)blockIdx.x * 256 + threadIdx.x;
  const int j    = (int)(i & 15);
  const int lane = (int)((i >> 4) & 31);
  const long tile = i >> 9;
  const int ntn = Nn >> 4;
  const int nt = (int)(tile % ntn);
  const int kt = (int)(tile / ntn);
  const int k = kt * 32 + ((lane >> 4) << 4) + j;
  const int n = nt * 16 + (lane & 15);
  dst[i] = (__bf16)(float)src[(long)k * Nn + n];
}

// ---------------- QKV projection: xn[8192x1024] @ Wpack -> bf16 Q/K/V ----------------
// One wave computes a 32(M) x 64(N) tile; ping-pong double-buffered fragments,
// 16 WMMAs per 64-K macro-step.
__global__ void __launch_bounds__(256) qkv_gemm_kernel(const __bf16* __restrict__ xn,
                                                       const __bf16* __restrict__ wpack,
                                                       const float* __restrict__ bqp,
                                                       const float* __restrict__ bkp,
                                                       const float* __restrict__ bvp,
                                                       __bf16* __restrict__ qkv) {
  const int lane = threadIdx.x & 31;
  const long waveId = (long)blockIdx.x * 8 + (threadIdx.x >> 5);
  const int MT = ROWS / 32, NT = HID / 64, per = MT * NT;   // 256 * 16
  const int which = (int)(waveId / per);
  const int rem   = (int)(waveId % per);
  const int mt = rem / NT, nt = rem % NT;
  const __bf16* Bp = wpack + (long)which * DIN * HID;
  const float* bias = (which == 0) ? bqp : (which == 1 ? bkp : bvp);
  __bf16* outp = qkv + (long)which * ROWS * HID;
  const int m0 = mt * 32, n0 = nt * 64;
  const int ntn = HID >> 4;
  __builtin_prefetch(xn + (long)m0 * DIN, 0, 3);
  __builtin_prefetch(Bp + ((long)(n0 >> 4) * 32) * 16, 0, 3);
  v8f acc[8] = {};
  // set A fragments @ k=0
  v16bf a0A = load_a_frag(xn, DIN, m0, 0, lane);
  v16bf a1A = load_a_frag(xn, DIN, m0 + 16, 0, lane);
  v16bf bA[4];
#pragma unroll
  for (int ni = 0; ni < 4; ++ni)
    bA[ni] = load_b_frag_packed(Bp, ntn, 0, (n0 >> 4) + ni, lane);
  for (int k0 = 0; k0 < DIN; k0 += 64) {
    // issue set B loads @ k0+32, then compute with set A
    const int kb = k0 + 32;
    v16bf a0B = load_a_frag(xn, DIN, m0, kb, lane);
    v16bf a1B = load_a_frag(xn, DIN, m0 + 16, kb, lane);
    v16bf bB[4];
#pragma unroll
    for (int ni = 0; ni < 4; ++ni)
      bB[ni] = load_b_frag_packed(Bp, ntn, kb >> 5, (n0 >> 4) + ni, lane);
#pragma unroll
    for (int ni = 0; ni < 4; ++ni) acc[ni] = wmma_bf16(a0A, bA[ni], acc[ni]);
#pragma unroll
    for (int ni = 0; ni < 4; ++ni) acc[4 + ni] = wmma_bf16(a1A, bA[ni], acc[4 + ni]);
    // issue set A loads @ k0+64 (wrap on last iter; values unused), compute with set B
    const int ka = (k0 + 64 < DIN) ? (k0 + 64) : 0;
    a0A = load_a_frag(xn, DIN, m0, ka, lane);
    a1A = load_a_frag(xn, DIN, m0 + 16, ka, lane);
#pragma unroll
    for (int ni = 0; ni < 4; ++ni)
      bA[ni] = load_b_frag_packed(Bp, ntn, ka >> 5, (n0 >> 4) + ni, lane);
#pragma unroll
    for (int ni = 0; ni < 4; ++ni) acc[ni] = wmma_bf16(a0B, bB[ni], acc[ni]);
#pragma unroll
    for (int ni = 0; ni < 4; ++ni) acc[4 + ni] = wmma_bf16(a1B, bB[ni], acc[4 + ni]);
  }
  const int cm = (lane >> 4) << 3;
  const int cn = lane & 15;
#pragma unroll
  for (int mi = 0; mi < 2; ++mi)
#pragma unroll
    for (int ni = 0; ni < 4; ++ni) {
      const float bb = bias[n0 + ni * 16 + cn];
#pragma unroll
      for (int r = 0; r < 8; ++r)
        outp[(long)(m0 + mi * 16 + cm + r) * HID + n0 + ni * 16 + cn] =
            (__bf16)(acc[mi * 4 + ni][r] + bb);
    }
}

// ---------------- Scores: S = (Q K^T) * scale, f32 straight into d_out ----------------
// One wave computes a 32(q) x 64(k) tile of S; ping-pong double-buffered fragments.
__global__ void __launch_bounds__(256) qk_gemm_kernel(const __bf16* __restrict__ qm,
                                                      const __bf16* __restrict__ km,
                                                      float* __restrict__ att) {
  const int lane = threadIdx.x & 31;
  const long waveId = (long)blockIdx.x * 8 + (threadIdx.x >> 5);
  const int MT = SEQ / 32, NT = SEQ / 64, per = MT * NT;    // 64 * 32
  const int b   = (int)(waveId / per);
  const int rem = (int)(waveId % per);
  const int mt = rem / NT, nt = rem % NT;
  const __bf16* Q = qm + (long)b * SEQ * HID;
  const __bf16* K = km + (long)b * SEQ * HID;
  float* S = att + (long)b * SEQ * SEQ;
  const int m0 = mt * 32, n0 = nt * 64;
  __builtin_prefetch(Q + (long)m0 * HID, 0, 3);
  __builtin_prefetch(K + (long)n0 * HID, 0, 3);
  v8f acc[8] = {};
  v16bf a0A = load_a_frag(Q, HID, m0, 0, lane);
  v16bf a1A = load_a_frag(Q, HID, m0 + 16, 0, lane);
  v16bf bA[4];
#pragma unroll
  for (int ni = 0; ni < 4; ++ni)
    bA[ni] = load_b_frag_rows(K, HID, n0 + ni * 16, 0, lane);
  for (int k0 = 0; k0 < HID; k0 += 64) {
    const int kb = k0 + 32;
    v16bf a0B = load_a_frag(Q, HID, m0, kb, lane);
    v16bf a1B = load_a_frag(Q, HID, m0 + 16, kb, lane);
    v16bf bB[4];
#pragma unroll
    for (int ni = 0; ni < 4; ++ni)
      bB[ni] = load_b_frag_rows(K, HID, n0 + ni * 16, kb, lane);
#pragma unroll
    for (int ni = 0; ni < 4; ++ni) acc[ni] = wmma_bf16(a0A, bA[ni], acc[ni]);
#pragma unroll
    for (int ni = 0; ni < 4; ++ni) acc[4 + ni] = wmma_bf16(a1A, bA[ni], acc[4 + ni]);
    const int ka = (k0 + 64 < HID) ? (k0 + 64) : 0;
    a0A = load_a_frag(Q, HID, m0, ka, lane);
    a1A = load_a_frag(Q, HID, m0 + 16, ka, lane);
#pragma unroll
    for (int ni = 0; ni < 4; ++ni)
      bA[ni] = load_b_frag_rows(K, HID, n0 + ni * 16, ka, lane);
#pragma unroll
    for (int ni = 0; ni < 4; ++ni) acc[ni] = wmma_bf16(a0B, bB[ni], acc[ni]);
#pragma unroll
    for (int ni = 0; ni < 4; ++ni) acc[4 + ni] = wmma_bf16(a1B, bB[ni], acc[4 + ni]);
  }
  const float scale = 0.03125f;  // 1/sqrt(1024)
  const int cm = (lane >> 4) << 3, cn = lane & 15;
#pragma unroll
  for (int mi = 0; mi < 2; ++mi)
#pragma unroll
    for (int ni = 0; ni < 4; ++ni)
#pragma unroll
      for (int r = 0; r < 8; ++r)
        S[(long)(m0 + mi * 16 + cm + r) * SEQ + n0 + ni * 16 + cn] =
            acc[mi * 4 + ni][r] * scale;
}

// ---------------- Row softmax over 2048; f32 in-place + bf16 probs for next GEMM ----------------
__global__ void __launch_bounds__(256) softmax_kernel(float* __restrict__ att,
                                                      __bf16* __restrict__ attp) {
  __shared__ float red[256];
  const long row = blockIdx.x;
  float* p  = att  + row * SEQ;
  __bf16* pp = attp + row * SEQ;
  const int t = threadIdx.x;
  float v[8];
  float m = -3.0e38f;
#pragma unroll
  for (int i = 0; i < 8; ++i) { v[i] = p[t + 256 * i]; m = fmaxf(m, v[i]); }
  red[t] = m; __syncthreads();
  for (int o = 128; o > 0; o >>= 1) { if (t < o) red[t] = fmaxf(red[t], red[t + o]); __syncthreads(); }
  m = red[0];
  __syncthreads();
  float s = 0.f;
#pragma unroll
  for (int i = 0; i < 8; ++i) { v[i] = __expf(v[i] - m); s += v[i]; }
  red[t] = s; __syncthreads();
  for (int o = 128; o > 0; o >>= 1) { if (t < o) red[t] += red[t + o]; __syncthreads(); }
  const float inv = 1.0f / red[0];
#pragma unroll
  for (int i = 0; i < 8; ++i) {
    const float o = v[i] * inv;
    p[t + 256 * i]  = o;
    pp[t + 256 * i] = (__bf16)o;
  }
}

// ---------------- out = probs @ V (V pre-packed), f32 into d_out ----------------
// One wave computes a 32(q) x 64(h) tile of out; ping-pong double-buffered fragments.
__global__ void __launch_bounds__(256) av_gemm_kernel(const __bf16* __restrict__ attp,
                                                      const __bf16* __restrict__ vpack,
                                                      float* __restrict__ outp) {
  const int lane = threadIdx.x & 31;
  const long waveId = (long)blockIdx.x * 8 + (threadIdx.x >> 5);
  const int MT = SEQ / 32, NT = HID / 64, per = MT * NT;    // 64 * 16
  const int b   = (int)(waveId / per);
  const int rem = (int)(waveId % per);
  const int mt = rem / NT, nt = rem % NT;
  const __bf16* A  = attp  + (long)b * SEQ * SEQ;
  const __bf16* Bp = vpack + (long)b * SEQ * HID;
  float* C = outp + (long)b * SEQ * HID;
  const int m0 = mt * 32, n0 = nt * 64;
  const int ntn = HID >> 4;
  __builtin_prefetch(A + (long)m0 * SEQ, 0, 3);
  __builtin_prefetch(Bp + ((long)(n0 >> 4) * 32) * 16, 0, 3);
  v8f acc[8] = {};
  v16bf a0A = load_a_frag(A, SEQ, m0, 0, lane);
  v16bf a1A = load_a_frag(A, SEQ, m0 + 16, 0, lane);
  v16bf bA[4];
#pragma unroll
  for (int ni = 0; ni < 4; ++ni)
    bA[ni] = load_b_frag_packed(Bp, ntn, 0, (n0 >> 4) + ni, lane);
  for (int k0 = 0; k0 < SEQ; k0 += 64) {
    const int kb = k0 + 32;
    v16bf a0B = load_a_frag(A, SEQ, m0, kb, lane);
    v16bf a1B = load_a_frag(A, SEQ, m0 + 16, kb, lane);
    v16bf bB[4];
#pragma unroll
    for (int ni = 0; ni < 4; ++ni)
      bB[ni] = load_b_frag_packed(Bp, ntn, kb >> 5, (n0 >> 4) + ni, lane);
#pragma unroll
    for (int ni = 0; ni < 4; ++ni) acc[ni] = wmma_bf16(a0A, bA[ni], acc[ni]);
#pragma unroll
    for (int ni = 0; ni < 4; ++ni) acc[4 + ni] = wmma_bf16(a1A, bA[ni], acc[4 + ni]);
    const int ka = (k0 + 64 < SEQ) ? (k0 + 64) : 0;
    a0A = load_a_frag(A, SEQ, m0, ka, lane);
    a1A = load_a_frag(A, SEQ, m0 + 16, ka, lane);
#pragma unroll
    for (int ni = 0; ni < 4; ++ni)
      bA[ni] = load_b_frag_packed(Bp, ntn, ka >> 5, (n0 >> 4) + ni, lane);
#pragma unroll
    for (int ni = 0; ni < 4; ++ni) acc[ni] = wmma_bf16(a0B, bB[ni], acc[ni]);
#pragma unroll
    for (int ni = 0; ni < 4; ++ni) acc[4 + ni] = wmma_bf16(a1B, bB[ni], acc[4 + ni]);
  }
  const int cm = (lane >> 4) << 3, cn = lane & 15;
#pragma unroll
  for (int mi = 0; mi < 2; ++mi)
#pragma unroll
    for (int ni = 0; ni < 4; ++ni)
#pragma unroll
      for (int r = 0; r < 8; ++r)
        C[(long)(m0 + mi * 16 + cm + r) * HID + n0 + ni * 16 + cn] = acc[mi * 4 + ni][r];
}

extern "C" void kernel_launch(void* const* d_in, const int* in_sizes, int n_in,
                              void* d_out, int out_size, void* d_ws, size_t ws_size,
                              hipStream_t stream) {
  (void)in_sizes; (void)n_in; (void)out_size; (void)ws_size;
  const float* x   = (const float*)d_in[0];
  const float* wq  = (const float*)d_in[1];
  const float* bq  = (const float*)d_in[2];
  const float* wk  = (const float*)d_in[3];
  const float* bk  = (const float*)d_in[4];
  const float* wv  = (const float*)d_in[5];
  const float* bv  = (const float*)d_in[6];
  const float* lnw = (const float*)d_in[7];
  const float* lnb = (const float*)d_in[8];
  float* out = (float*)d_out;

  // Workspace layout (bytes, all 32B aligned):
  char* ws = (char*)d_ws;
  __bf16* xn    = (__bf16*)(ws);                         // 16 MB
  __bf16* wpack = (__bf16*)(ws + 16777216);              //  6 MB (3x 1024x1024 packed)
  __bf16* qkv   = (__bf16*)(ws + 23068672);              // 48 MB (Q,K,V row-major bf16)
  __bf16* qb = qkv;
  __bf16* kb = qkv + (long)ROWS * HID;
  __bf16* vb = qkv + 2L * ROWS * HID;
  __bf16* vpack = (__bf16*)(ws + 73400320);              // 16 MB
  __bf16* attp  = (__bf16*)(ws + 90177536);              // 32 MB
  float* att  = out;                                     // [4,2048,2048] f32
  float* outp = out + (long)BATCH * SEQ * SEQ;           // [4,2048,1024] f32

  ln_kernel<<<ROWS, 256, 0, stream>>>(x, lnw, lnb, xn);

  pack_b_kernel<float><<<dim3(DIN * HID / 256, 1), 256, 0, stream>>>(
      wq, wpack, HID, 0, 0);
  pack_b_kernel<float><<<dim3(DIN * HID / 256, 1), 256, 0, stream>>>(
      wk, wpack + (long)DIN * HID, HID, 0, 0);
  pack_b_kernel<float><<<dim3(DIN * HID / 256, 1), 256, 0, stream>>>(
      wv, wpack + 2L * DIN * HID, HID, 0, 0);

  // 3 * (8192/32) * (1024/64) = 12288 waves -> 1536 blocks
  qkv_gemm_kernel<<<3 * (ROWS / 32) * (HID / 64) / 8, 256, 0, stream>>>(
      xn, wpack, bq, bk, bv, qkv);

  pack_b_kernel<__bf16><<<dim3(SEQ * HID / 256, BATCH), 256, 0, stream>>>(
      vb, vpack, HID, (long)SEQ * HID, (long)SEQ * HID);

  // 4 * (2048/32) * (2048/64) = 8192 waves -> 1024 blocks
  qk_gemm_kernel<<<BATCH * (SEQ / 32) * (SEQ / 64) / 8, 256, 0, stream>>>(qb, kb, att);

  softmax_kernel<<<ROWS, 256, 0, stream>>>(att, attp);

  // 4 * (2048/32) * (1024/64) = 4096 waves -> 512 blocks
  av_gemm_kernel<<<BATCH * (SEQ / 32) * (HID / 64) / 8, 256, 0, stream>>>(
      attp, vpack, outp);
}
